// MultiHeadSelfAttention_40157944218157
// MI455X (gfx1250) — matrix operand order repaired and checked by reading, not compiled, not run on hardware
//
#include <hip/hip_runtime.h>
#include <hip/hip_bf16.h>

typedef __bf16 bf16;
typedef __attribute__((ext_vector_type(16))) __bf16 v16bf;
typedef __attribute__((ext_vector_type(8)))  __bf16 v8bf;
typedef __attribute__((ext_vector_type(8)))  float  v8f;

union ABf { v16bf v; v8bf h[2]; };

static constexpr int S = 4096;   // sequence
static constexpr int E = 768;    // embedding
static constexpr int H = 12;     // heads
static constexpr int D = 64;     // head dim

__device__ __forceinline__ v8f wmma_bf16(v16bf a, v16bf b, v8f c) {
    return __builtin_amdgcn_wmma_f32_16x16x32_bf16(
        /*neg_a=*/false, a, /*neg_b=*/false, b,
        /*c_mod=*/(short)0, c, /*reuse_a=*/false, /*reuse_b=*/false);
}

// lane <-> lane^16 exchange, pure VALU (v_permlanex16_b32 with identity selector)
__device__ __forceinline__ float permx16(float x) {
    int i = __builtin_bit_cast(int, x);
    int r = __builtin_amdgcn_permlanex16(i, i, 0x76543210u, 0xfedcba98u, false, false);
    return __builtin_bit_cast(float, r);
}

// ---------------- conversion kernels ----------------
__global__ void cvt_bf16(const float* __restrict__ src, bf16* __restrict__ dst, int n) {
    int i = blockIdx.x * blockDim.x + threadIdx.x;
    if (i < n) dst[i] = (bf16)src[i];
}

// W [K x N] row-major f32  ->  Wt [N x K] row-major bf16
__global__ void transpose_bf16(const float* __restrict__ src, bf16* __restrict__ dst,
                               int K, int N) {
    int i = blockIdx.x * blockDim.x + threadIdx.x;
    if (i < K * N) {
        int k = i / N, n = i % N;
        dst[n * K + k] = (bf16)src[i];
    }
}

// ---------------- GEMM:  C[M x N] = A[M x K] * Bt[N x K]^T + bias ----------------
// MODE 0: store bf16 row-major [m][n]
// MODE 1: store bf16 transposed [n*M + m]   (used to build V^T per head)
// MODE 2: store f32 row-major [m][n]
template <int MODE>
__global__ void gemm_bf16(const bf16* __restrict__ A, const bf16* __restrict__ Bt,
                          const float* __restrict__ bias, void* __restrict__ Cout,
                          int M, int N, int Kdim) {
    const int lane = threadIdx.x & 31;
    const int w    = threadIdx.x >> 5;
    const int g    = lane >> 4;      // lane half
    const int ln   = lane & 15;
    const int wr   = w & 3;          // 4 row strips
    const int wc   = w >> 2;         // 2 col strips
    const int m0   = blockIdx.x * 128 + wr * 32;
    const int n0   = blockIdx.y * 128 + wc * 64;

    v8f acc[2][4] = {};

    const bf16* arow0 = A + (size_t)(m0 + ln) * Kdim;
    const bf16* arow1 = A + (size_t)(m0 + 16 + ln) * Kdim;

    for (int k0 = 0; k0 < Kdim; k0 += 32) {
        ABf a0, a1;
        a0.h[0] = *(const v8bf*)(arow0 + k0 + g * 8);
        a0.h[1] = *(const v8bf*)(arow0 + k0 + 16 + g * 8);
        a1.h[0] = *(const v8bf*)(arow1 + k0 + g * 8);
        a1.h[1] = *(const v8bf*)(arow1 + k0 + 16 + g * 8);
#pragma unroll
        for (int c = 0; c < 4; ++c) {
            ABf b;
            b.h[0] = *(const v8bf*)(Bt + (size_t)(n0 + c * 16 + ln) * Kdim + k0 + g * 8);
            b.h[1] = *(const v8bf*)(Bt + (size_t)(n0 + c * 16 + ln) * Kdim + k0 + 16 + g * 8);
            acc[0][c] = wmma_bf16(a0.v, b.v, acc[0][c]);
            acc[1][c] = wmma_bf16(a1.v, b.v, acc[1][c]);
        }
    }

#pragma unroll
    for (int c = 0; c < 4; ++c) {
        const int n  = n0 + c * 16 + ln;
        const float bv = bias[n];
#pragma unroll
        for (int t = 0; t < 2; ++t) {
#pragma unroll
            for (int r = 0; r < 8; ++r) {
                const int m   = m0 + t * 16 + r + 8 * g;
                const float v = acc[t][c][r] + bv;
                if (MODE == 0)      ((bf16*)Cout)[(size_t)m * N + n] = (bf16)v;
                else if (MODE == 1) ((bf16*)Cout)[(size_t)n * M + m] = (bf16)v;
                else                ((float*)Cout)[(size_t)m * N + n] = v;
            }
        }
    }
}

// ---------------- Flash attention (transposed: S^T = K Q^T, O^T = V^T P^T) ------
// Q, K : bf16 [S][E] (head h = cols h*64..h*64+63), Vt : bf16 [E][S] (row h*64+d)
// Out  : bf16 [S][E]
// Per wave: 32 queries (2 query tiles sharing K / V^T fragments).
// Lane layout: query = lane&15 everywhere; key/d = register index.
// Needs ~200 VGPRs: relax occupancy so the allocator doesn't spill.
__global__ void
__launch_bounds__(128)
__attribute__((amdgpu_waves_per_eu(1, 4)))
flash_attn(const bf16* __restrict__ Q, const bf16* __restrict__ K,
           const bf16* __restrict__ Vt, bf16* __restrict__ O) {
    const int lane = threadIdx.x & 31;
    const int w    = threadIdx.x >> 5;   // 4 waves
    const int g    = lane >> 4;
    const int ln   = lane & 15;
    const bool hic = (g != 0);
    const int h    = blockIdx.y;
    const int q0   = blockIdx.x * 128 + w * 32;

    // Q^T B-fragments (d x 16 queries) per query tile: element i <-> d = half*32 + (i/8)*16 + g*8 + i%8
    v16bf qb[2][2];
#pragma unroll
    for (int qt = 0; qt < 2; ++qt) {
        const bf16* qrow = Q + (size_t)(q0 + qt * 16 + ln) * E + h * D;
        ABf t0, t1;
        t0.h[0] = *(const v8bf*)(qrow + g * 8);
        t0.h[1] = *(const v8bf*)(qrow + 16 + g * 8);
        t1.h[0] = *(const v8bf*)(qrow + 32 + g * 8);
        t1.h[1] = *(const v8bf*)(qrow + 48 + g * 8);
        qb[qt][0] = t0.v;
        qb[qt][1] = t1.v;
    }

    float mrun[2] = {-1.0e30f, -1.0e30f};
    float lrun[2] = {0.0f, 0.0f};
    v8f o[2][4] = {};   // O^T tiles per query tile: 16 d-rows x 16 queries

    for (int kb = 0; kb < S; kb += 32) {
        // K A-fragments (shared by both query tiles):
        // 16 keys x 32 d, element i <-> d = half*32 + (i/8)*16 + g*8 + i%8
        const bf16* kr0 = K + (size_t)(kb + ln) * E + h * D;
        const bf16* kr1 = K + (size_t)(kb + 16 + ln) * E + h * D;
        ABf ka00, ka01, ka10, ka11;
        ka00.h[0] = *(const v8bf*)(kr0 + g * 8);
        ka00.h[1] = *(const v8bf*)(kr0 + 16 + g * 8);
        ka01.h[0] = *(const v8bf*)(kr0 + 32 + g * 8);
        ka01.h[1] = *(const v8bf*)(kr0 + 48 + g * 8);
        ka10.h[0] = *(const v8bf*)(kr1 + g * 8);
        ka10.h[1] = *(const v8bf*)(kr1 + 16 + g * 8);
        ka11.h[0] = *(const v8bf*)(kr1 + 32 + g * 8);
        ka11.h[1] = *(const v8bf*)(kr1 + 48 + g * 8);

        v16bf pb[2];
        float alpha[2];
#pragma unroll
        for (int qt = 0; qt < 2; ++qt) {
            // S^T tiles: st0 = keys kb..kb+15, st1 = keys kb+16..kb+31
            v8f st0 = {}, st1 = {};
            st0 = wmma_bf16(ka00.v, qb[qt][0], st0);
            st0 = wmma_bf16(ka01.v, qb[qt][1], st0);
            st1 = wmma_bf16(ka10.v, qb[qt][0], st1);
            st1 = wmma_bf16(ka11.v, qb[qt][1], st1);

            // ---- online softmax: per-query stats live entirely in-lane ----
            // max commutes with the positive 1/sqrt(d) scale: reduce raw, scale once.
            float mx = fmaxf(st0[0], st1[0]);
#pragma unroll
            for (int r = 1; r < 8; ++r) mx = fmaxf(mx, fmaxf(st0[r], st1[r]));
            mx *= 0.125f;
            mx = fmaxf(mx, permx16(mx));           // combine the two key-halves

            const float mnew = fmaxf(mrun[qt], mx);
            alpha[qt] = __expf(mrun[qt] - mnew);

            float p0[8], p1[8], lsum = 0.0f;
#pragma unroll
            for (int r = 0; r < 8; ++r) {
                p0[r] = __expf(fmaf(st0[r], 0.125f, -mnew));
                p1[r] = __expf(fmaf(st1[r], 0.125f, -mnew));
                lsum += p0[r] + p1[r];
            }
            lsum += permx16(lsum);                 // combine the two key-halves
            lrun[qt] = lrun[qt] * alpha[qt] + lsum;
            mrun[qt] = mnew;

            // ---- build P^T B-fragment (32 keys x 16 queries), one cross-half swap ----
            // element i (lane half g): key = g*16 + i.
#pragma unroll
            for (int j = 0; j < 8; ++j) {
                float t  = hic ? p0[j] : p1[j];
                float tw = permx16(t);
                pb[qt][j]     = (bf16)(hic ? tw : p0[j]);
                pb[qt][8 + j] = (bf16)(hic ? p1[j] : tw);
            }
        }

#pragma unroll
        for (int qt = 0; qt < 2; ++qt)
#pragma unroll
            for (int c = 0; c < 4; ++c) o[qt][c] *= alpha[qt];

        // ---- O^T += V^T * P^T (V^T fragments shared by both query tiles) ----
#pragma unroll
        for (int c = 0; c < 4; ++c) {
            const bf16* vr = Vt + (size_t)(h * D + c * 16 + ln) * S + kb;
            ABf va;
            va.v = *(const v16bf*)(vr + g * 16);
            o[0][c] = wmma_bf16(va.v, pb[0], o[0][c]);
            o[1][c] = wmma_bf16(va.v, pb[1], o[1][c]);
        }
    }

    // normalize + vectorized store: lane writes 8 consecutive d values per tile
#pragma unroll
    for (int qt = 0; qt < 2; ++qt) {
        const float rl = 1.0f / lrun[qt];
#pragma unroll
        for (int c = 0; c < 4; ++c) {
            v8bf ov;
#pragma unroll
            for (int r = 0; r < 8; ++r) ov[r] = (bf16)(o[qt][c][r] * rl);
            *(v8bf*)(O + (size_t)(q0 + qt * 16 + ln) * E + h * D + c * 16 + 8 * g) = ov;
        }
    }
}

// ---------------- host launch ----------------
extern "C" void kernel_launch(void* const* d_in, const int* in_sizes, int n_in,
                              void* d_out, int out_size, void* d_ws, size_t ws_size,
                              hipStream_t stream) {
    const float* x  = (const float*)d_in[0];
    const float* Wk = (const float*)d_in[1];
    const float* bk = (const float*)d_in[2];
    const float* Wq = (const float*)d_in[3];
    const float* bq = (const float*)d_in[4];
    const float* Wv = (const float*)d_in[5];
    const float* bv = (const float*)d_in[6];
    const float* Wo = (const float*)d_in[7];
    const float* bo = (const float*)d_in[8];

    char* ws = (char*)d_ws;
    size_t off = 0;
    auto alloc = [&](size_t bytes) -> void* {
        void* p = ws + off;
        off += (bytes + 255) & ~(size_t)255;
        return p;
    };
    bf16* xb  = (bf16*)alloc((size_t)S * E * 2);
    bf16* Qb  = (bf16*)alloc((size_t)S * E * 2);
    bf16* Kb  = (bf16*)alloc((size_t)S * E * 2);
    bf16* Vt  = (bf16*)alloc((size_t)S * E * 2);
    bf16* At  = (bf16*)alloc((size_t)S * E * 2);
    bf16* WqT = (bf16*)alloc((size_t)E * E * 2);
    bf16* WkT = (bf16*)alloc((size_t)E * E * 2);
    bf16* WvT = (bf16*)alloc((size_t)E * E * 2);
    bf16* WoT = (bf16*)alloc((size_t)E * E * 2);

    // conversions
    cvt_bf16<<<(S * E) / 256, 256, 0, stream>>>(x, xb, S * E);
    transpose_bf16<<<(E * E) / 256, 256, 0, stream>>>(Wq, WqT, E, E);
    transpose_bf16<<<(E * E) / 256, 256, 0, stream>>>(Wk, WkT, E, E);
    transpose_bf16<<<(E * E) / 256, 256, 0, stream>>>(Wv, WvT, E, E);
    transpose_bf16<<<(E * E) / 256, 256, 0, stream>>>(Wo, WoT, E, E);

    dim3 ggrid(S / 128, E / 128);     // 32 x 6
    gemm_bf16<0><<<ggrid, 256, 0, stream>>>(xb, WqT, bq, (void*)Qb, S, E, E);
    gemm_bf16<0><<<ggrid, 256, 0, stream>>>(xb, WkT, bk, (void*)Kb, S, E, E);
    gemm_bf16<1><<<ggrid, 256, 0, stream>>>(xb, WvT, bv, (void*)Vt, S, E, E);

    flash_attn<<<dim3(S / 128, H), 128, 0, stream>>>(Qb, Kb, Vt, At);

    gemm_bf16<2><<<ggrid, 256, 0, stream>>>(At, WoT, bo, d_out, S, E, E);
}
